// Head_91044716740856
// MI455X (gfx1250) — compile-verified, matrix operand order checked by reading, not run on hardware
//
#include <hip/hip_runtime.h>
#include <hip/hip_bf16.h>

#define EMBED 1024
#define HEAD  128
#define BB    4
#define TT    4096
#define QROWS 64   // query rows per attention block (4 waves x 16)

typedef __attribute__((ext_vector_type(16))) __bf16 v16bf;
typedef __attribute__((ext_vector_type(8)))  float  v8f;
typedef __attribute__((ext_vector_type(4)))  unsigned int u32x4;
typedef __attribute__((ext_vector_type(8)))  int  i32x8;
typedef __attribute__((ext_vector_type(4)))  int  i32x4;

static __device__ __forceinline__ v8f wmma_bf16(v16bf a, v16bf b, v8f c) {
  // (neg_a, A, neg_b, B, c_mod, C, reuse_a, reuse_b)
  return __builtin_amdgcn_wmma_f32_16x16x32_bf16(false, a, false, b, (short)0, c, false, false);
}

// Fragment layout (CDNA5 ISA): lane half hf = lane>>4;
// element e: K = hf*8+e (e<8), else 16+hf*8+(e-8). So bytes 0..15 of the
// v16bf are the first 16B load, bytes 16..31 the second -> pure bit-cast.
union Frag { v16bf v; uint4 q[2]; };

static __device__ __forceinline__ v16bf load_frag(const __bf16* base, int row, int stride, int hf) {
  const __bf16* p = base + (size_t)row * stride + hf * 8;
  Frag f;
  f.q[0] = *(const uint4*)(p);
  f.q[1] = *(const uint4*)(p + 16);
  return f.v;
}

// ---------------------------------------------------------------------------
// Tensor Data Mover: 2D bf16 tile, global -> LDS (CDNA5 ISA)
// ---------------------------------------------------------------------------
#if defined(__has_builtin)
#if __has_builtin(__builtin_amdgcn_tensor_load_to_lds)
#define HAVE_TDM 1
#endif
#endif

static __device__ __forceinline__ unsigned lds_off(const void* p) {
  return (unsigned)(unsigned long long)p;  // flat LDS addr: low 32 bits = LDS offset
}

#ifdef HAVE_TDM
static __device__ __forceinline__ void tdm_load_2d_bf16(
    unsigned lds_addr, const void* gaddr,
    unsigned tile_d0, unsigned tile_d1,     // tile: row length, #rows (elements)
    unsigned tensor_d0, unsigned tensor_d1, // tensor dims (elements)
    unsigned stride_d0) {                   // elements between rows
  unsigned long long ga = (unsigned long long)gaddr;
  u32x4 g0;
  g0[0] = 1u;                                        // count=1, user mode, no gather
  g0[1] = lds_addr;                                  // [63:32]  lds_addr
  g0[2] = (unsigned)(ga & 0xffffffffu);              // [95:64]  global_addr lo
  g0[3] = (unsigned)((ga >> 32) & 0x01ffffffu)       // [120:96] global_addr hi
        | (2u << 30);                                // [127:126] type=2 (image)
  i32x8 g1;
  g1[0] = (int)(1u << 16);                           // data_size=1 -> 2 bytes
  g1[1] = (int)((tensor_d0 & 0xffffu) << 16);        // tensor_dim0[15:0] @bits 63:48
  g1[2] = (int)(((tensor_d0 >> 16) & 0xffffu)        // tensor_dim0[31:16]
        | ((tensor_d1 & 0xffffu) << 16));            // tensor_dim1[15:0]
  g1[3] = (int)(((tensor_d1 >> 16) & 0xffffu)        // tensor_dim1[31:16]
        | ((tile_d0 & 0xffffu) << 16));              // tile_dim0
  g1[4] = (int)(tile_d1 & 0xffffu);                  // tile_dim1 (tile_dim2=0)
  g1[5] = (int)stride_d0;                            // tensor_dim0_stride lo32
  g1[6] = 0;                                         // stride hi / dim1_stride lo
  g1[7] = 0;
  i32x4 z4 = {};
#if __clang_major__ >= 23
  i32x8 z8 = {};
  __builtin_amdgcn_tensor_load_to_lds(g0, g1, z4, z4, z8, 0);
#else
  __builtin_amdgcn_tensor_load_to_lds(g0, g1, z4, z4, 0);
#endif
}
#endif

// ---------------------------------------------------------------------------
// Kernel 1: Q/K/V projection.  out[16x128] = ctx[16x1024] @ W[1024x128], bf16 out.
// grid = (B*T/16, 3), block = 256 (8 waves; wave w owns N-tile w).
// ---------------------------------------------------------------------------
__global__ __launch_bounds__(256) void proj_kernel(
    const float* __restrict__ ctx, const float* __restrict__ Wq,
    const float* __restrict__ Wk, const float* __restrict__ Wv,
    __bf16* __restrict__ qb, __bf16* __restrict__ kb, __bf16* __restrict__ vb) {
  __shared__ __align__(16) __bf16 ctx_lds[16 * EMBED];  // 32 KB
  __shared__ __align__(16) __bf16 w_lds[HEAD * 32];     // 8 KB, transposed [n][d]

  const int tid   = threadIdx.x;
  const int which = blockIdx.y;  // 0=q, 1=k, 2=v
  const float* W  = (which == 0) ? Wq : (which == 1) ? Wk : Wv;
  __bf16* out     = (which == 0) ? qb : (which == 1) ? kb : vb;
  const long rowbase = (long)blockIdx.x * 16;

  // Stage context tile (fp32 -> bf16) once.
  for (int i = tid; i < 16 * EMBED; i += 256) {
    int r = i >> 10, c = i & (EMBED - 1);
    ctx_lds[i] = (__bf16)ctx[(rowbase + r) * EMBED + c];
  }
  __syncthreads();

  const int wave = tid >> 5, lane = tid & 31;
  const int hf = lane >> 4, ln = lane & 15;
  const int nt = wave;  // N-tile 0..7

  v8f c = {};
  for (int kt = 0; kt < EMBED / 32; ++kt) {
    // Stage W chunk transposed: w_lds[n][d], d = 0..31 within this K chunk.
    for (int i = tid; i < 32 * HEAD; i += 256) {
      int d = i >> 7, n = i & (HEAD - 1);
      w_lds[n * 32 + d] = (__bf16)W[(size_t)(kt * 32 + d) * HEAD + n];
    }
    __syncthreads();
    v16bf a = load_frag(ctx_lds + kt * 32, ln, EMBED, hf);       // A: ctx rows
    v16bf b = load_frag(w_lds, nt * 16 + ln, 32, hf);            // B: W columns
    c = wmma_bf16(a, b, c);
    __syncthreads();
  }

#pragma unroll
  for (int r = 0; r < 8; ++r) {
    int m = r + 8 * hf;  // C layout: M = r + 8*half, N = lane&15
    out[(rowbase + m) * HEAD + nt * 16 + ln] = (__bf16)c[r];
  }
}

// ---------------------------------------------------------------------------
// Kernel 2: causal flash attention.  block = 128 thr (4 waves), 64 query rows.
// K/V tiles fetched with the Tensor Data Mover; V transposed LDS->LDS.
// ---------------------------------------------------------------------------
__global__ __launch_bounds__(128) void attn_kernel(
    const __bf16* __restrict__ qb, const __bf16* __restrict__ kb,
    const __bf16* __restrict__ vb, float* __restrict__ out) {
  __shared__ __align__(16) __bf16 k_lds[32 * HEAD];        // 8 KB, [key][hd]
  __shared__ __align__(16) __bf16 v_raw[32 * HEAD];        // 8 KB, [key][hd]
  __shared__ __align__(16) __bf16 vT_lds[HEAD * 32];       // 8 KB, [hd][key]
  __shared__ __align__(16) __bf16 p_lds[4][16][32];        // 4 KB, per-wave P

  const int tid = threadIdx.x;
  const int wv = tid >> 5, lane = tid & 31, hf = lane >> 4, ln = lane & 15;
  const int blk = blockIdx.x;
  const int b  = blk / (TT / QROWS);
  const int q0 = (blk % (TT / QROWS)) * QROWS;
  const long base = (long)b * TT;
  const int qrow0 = q0 + wv * 16;

  // Q fragments for this wave: 4 K-chunks of 32 over the 128-wide head.
  v16bf qf[4];
#pragma unroll
  for (int kc = 0; kc < 4; ++kc)
    qf[kc] = load_frag(qb + (base + qrow0) * HEAD + kc * 32, ln, HEAD, hf);

  float mrow[8], lrow[8];
  v8f acc[8];
#pragma unroll
  for (int r = 0; r < 8; ++r) { mrow[r] = -1e30f; lrow[r] = 0.0f; }
#pragma unroll
  for (int nt = 0; nt < 8; ++nt) acc[nt] = (v8f){};

  // softmax in base-2: fold 1/sqrt(1024) * log2(e) into the score scale.
  const float sc = 0.03125f * 1.4426950408889634f;
  const int numj = (q0 + QROWS) / 32;  // block-uniform trip count

  for (int j = 0; j < numj; ++j) {
    const int kbase = j * 32;
#ifdef HAVE_TDM
    if (wv == 0) {  // TDM ignores EXEC; issue once per tile from wave 0
      tdm_load_2d_bf16(lds_off(k_lds), kb + (base + kbase) * HEAD,
                       HEAD, 32, HEAD, 32, HEAD);
      tdm_load_2d_bf16(lds_off(v_raw), vb + (base + kbase) * HEAD,
                       HEAD, 32, HEAD, 32, HEAD);
#if __has_builtin(__builtin_amdgcn_s_wait_tensorcnt)
      __builtin_amdgcn_s_wait_tensorcnt(0);
#endif
    }
#else
    {
      const uint4* gk = (const uint4*)(kb + (base + kbase) * HEAD);
      const uint4* gv = (const uint4*)(vb + (base + kbase) * HEAD);
      uint4* lk = (uint4*)k_lds;
      uint4* lv = (uint4*)v_raw;
      for (int i = tid; i < (32 * HEAD) / 8; i += 128) { lk[i] = gk[i]; lv[i] = gv[i]; }
    }
#endif
    if (j + 1 < numj) {  // gfx1250 global_prefetch_b8 for the next tiles
      __builtin_prefetch(kb + (base + kbase + 32) * HEAD, 0, 1);
      __builtin_prefetch(vb + (base + kbase + 32) * HEAD, 0, 1);
    }
    __syncthreads();  // K/V tiles resident in LDS

    // Transpose V in LDS: vT[hd][key] (feeds O = P@V B-fragments).
    for (int i = tid; i < 32 * HEAD; i += 128) {
      int d = i >> 7, hd = i & (HEAD - 1);
      vT_lds[hd * 32 + d] = v_raw[i];
    }

    // S = Q @ K^T for two 16-key column tiles.
    v8f s[2];
#pragma unroll
    for (int ncol = 0; ncol < 2; ++ncol) {
      v8f cf = {};
#pragma unroll
      for (int kc = 0; kc < 4; ++kc) {
        v16bf bf = load_frag(k_lds + kc * 32, ncol * 16 + ln, HEAD, hf);
        cf = wmma_bf16(qf[kc], bf, cf);
      }
      s[ncol] = cf;
    }

    // Scale + causal mask (C layout: M = r+8*hf, N = lane&15).
#pragma unroll
    for (int ncol = 0; ncol < 2; ++ncol)
#pragma unroll
      for (int r = 0; r < 8; ++r) {
        int qi = qrow0 + r + 8 * hf;
        int ki = kbase + ncol * 16 + ln;
        float v = s[ncol][r] * sc;
        s[ncol][r] = (ki <= qi) ? v : -1e30f;
      }

    // Online softmax: per-row stats; rows live across 16-lane halves.
#pragma unroll
    for (int r = 0; r < 8; ++r) {
      float mx = fmaxf(s[0][r], s[1][r]);
#pragma unroll
      for (int off = 1; off < 16; off <<= 1) mx = fmaxf(mx, __shfl_xor(mx, off, 32));
      float mnew = fmaxf(mrow[r], mx);
      float corr = exp2f(mrow[r] - mnew);
      mrow[r] = mnew;
      float p0 = exp2f(s[0][r] - mnew);
      float p1 = exp2f(s[1][r] - mnew);
      s[0][r] = p0; s[1][r] = p1;
      float ps = p0 + p1;
#pragma unroll
      for (int off = 1; off < 16; off <<= 1) ps += __shfl_xor(ps, off, 32);
      lrow[r] = lrow[r] * corr + ps;
#pragma unroll
      for (int nt = 0; nt < 8; ++nt) acc[nt][r] = acc[nt][r] * corr;
    }

    // C-layout -> A-layout for P via LDS (per-wave region).
#pragma unroll
    for (int ncol = 0; ncol < 2; ++ncol)
#pragma unroll
      for (int r = 0; r < 8; ++r)
        p_lds[wv][r + 8 * hf][ncol * 16 + ln] = (__bf16)s[ncol][r];
    __syncthreads();  // P + vT ready

    v16bf pf = load_frag(&p_lds[wv][0][0], ln, 32, hf);
#pragma unroll
    for (int nt = 0; nt < 8; ++nt) {
      v16bf bf = load_frag(vT_lds, nt * 16 + ln, 32, hf);  // B: V cols, keys contiguous
      acc[nt] = wmma_bf16(pf, bf, acc[nt]);
    }
    __syncthreads();  // all reads done before next tile overwrites LDS
  }

  // Normalize and store fp32 output.
#pragma unroll
  for (int r = 0; r < 8; ++r) {
    float inv = 1.0f / lrow[r];
    int qi = qrow0 + r + 8 * hf;
#pragma unroll
    for (int nt = 0; nt < 8; ++nt)
      out[(base + qi) * HEAD + nt * 16 + ln] = acc[nt][r] * inv;
  }
}

// ---------------------------------------------------------------------------
extern "C" void kernel_launch(void* const* d_in, const int* in_sizes, int n_in,
                              void* d_out, int out_size, void* d_ws, size_t ws_size,
                              hipStream_t stream) {
  const float* ctx = (const float*)d_in[0];
  const float* Wk  = (const float*)d_in[1];
  const float* Wq  = (const float*)d_in[2];
  const float* Wv  = (const float*)d_in[3];
  float* out = (float*)d_out;

  const size_t n = (size_t)BB * TT * HEAD;
  __bf16* qb = (__bf16*)d_ws;
  __bf16* kb = qb + n;
  __bf16* vb = kb + n;

  proj_kernel<<<dim3((BB * TT) / 16, 3), 256, 0, stream>>>(ctx, Wq, Wk, Wv, qb, kb, vb);
  attn_kernel<<<dim3(BB * (TT / QROWS)), 128, 0, stream>>>(qb, kb, vb, out);
}
